// EGNNLayer_893353197946
// MI455X (gfx1250) — matrix operand order repaired and checked by reading, not compiled, not run on hardware
//
#include <hip/hip_runtime.h>

#define NN 50000
#define NE 800000
#define DD 128
#define CLAMPV 2.0f

typedef __attribute__((ext_vector_type(16))) __bf16 v16bf;
typedef __attribute__((ext_vector_type(16))) short  v16s;
typedef __attribute__((ext_vector_type(8)))  short  v8s;
typedef __attribute__((ext_vector_type(8)))  float  v8f;

static __device__ __forceinline__ unsigned short f2bf(float f) {
    unsigned u = __builtin_bit_cast(unsigned, f);
    u += 0x7FFFu + ((u >> 16) & 1u);            // round-to-nearest-even
    return (unsigned short)(u >> 16);
}

static __device__ __forceinline__ float silu(float x) {
    return x / (1.0f + __expf(-x));
}

static __device__ __forceinline__ v8f wmma_bf16(v16s a, v16s b, v8f c) {
    return __builtin_amdgcn_wmma_f32_16x16x32_bf16(
        false, __builtin_bit_cast(v16bf, a),
        false, __builtin_bit_cast(v16bf, b),
        (short)0, c, false, false);
}

static __device__ __forceinline__ void ld8(const float* p, float* d) {
    float4 a = *(const float4*)p;
    float4 b = *(const float4*)(p + 4);
    d[0] = a.x; d[1] = a.y; d[2] = a.z; d[3] = a.w;
    d[4] = b.x; d[5] = b.y; d[6] = b.z; d[7] = b.w;
}

// A-fragment (16x32 bf16) from a bf16 row: lane (n15,kh) -> M=n15,
// elements 0..7 : K = 32f + kh*8 + e ; 8..15 : K = 32f + 16 + kh*8 + e
static __device__ __forceinline__ v16s afrag_bf(const unsigned short* row, int f, int kh) {
    int k0 = f * 32 + kh * 8;
    v8s lo = *(const v8s*)(row + k0);
    v8s hi = *(const v8s*)(row + k0 + 16);
    v16s a;
#pragma unroll
    for (int i = 0; i < 8; i++) { a[i] = lo[i]; a[8 + i] = hi[i]; }
    return a;
}

// B-fragment (32x16 bf16) from transposed weight WT[N][K]: lane -> N = nt*16+n15,
// elements 0..7 : K = 32f + kh*8 + e ; 8..15 : K = 32f + 16 + kh*8 + e
static __device__ __forceinline__ v16s bfrag(const unsigned short* wTmat, int nt, int n15, int f, int kh) {
    const unsigned short* p = wTmat + (size_t)(nt * 16 + n15) * DD + f * 32 + kh * 8;
    v8s lo = *(const v8s*)p;
    v8s hi = *(const v8s*)(p + 16);
    v16s b;
#pragma unroll
    for (int i = 0; i < 8; i++) { b[i] = lo[i]; b[8 + i] = hi[i]; }
    return b;
}

// ---------------- prep: bf16 copy of h, zero aggregation buffers ----------------
__global__ void k_prep(const float* __restrict__ h, unsigned short* __restrict__ hb,
                       float* __restrict__ agg_m, float* __restrict__ agg_x,
                       float* __restrict__ cnt) {
    int i = blockIdx.x * blockDim.x + threadIdx.x;
    if (i < NN * DD) { hb[i] = f2bf(h[i]); agg_m[i] = 0.f; }
    if (i < NN * 3) agg_x[i] = 0.f;
    if (i < NN) cnt[i] = 0.f;
}

// ------------- transpose + convert the 7 128x128 weight blocks to bf16 -------------
__global__ void k_wT(const float* __restrict__ we1, const float* __restrict__ we2,
                     const float* __restrict__ wc1, const float* __restrict__ wn1,
                     const float* __restrict__ wn2, unsigned short* __restrict__ wT) {
    int m = blockIdx.y;
    int i = blockIdx.x * blockDim.x + threadIdx.x;  // 0..16383
    int n = i >> 7, k = i & 127;
    const float* src = we1; int rowoff = 0;
    switch (m) {
        case 0: src = we1; rowoff = 0;   break;  // we1 rows   0..127
        case 1: src = we1; rowoff = 128; break;  // we1 rows 128..255
        case 2: src = we2; break;
        case 3: src = wc1; break;
        case 4: src = wn1; rowoff = 0;   break;  // wn1 rows   0..127
        case 5: src = wn1; rowoff = 128; break;  // wn1 rows 128..255
        default: src = wn2; break;
    }
    wT[(size_t)m * 16384 + n * 128 + k] = f2bf(src[(size_t)(rowoff + k) * 128 + n]);
}

// ------------- per-node precompute: hA = h@we1_top + be1, hB = h@we1_bot -------------
__global__ __launch_bounds__(256) void k_hAhB(const unsigned short* __restrict__ hb,
                                              const unsigned short* __restrict__ wT,
                                              const float* __restrict__ be1,
                                              float* __restrict__ hA, float* __restrict__ hB) {
    int wave = threadIdx.x >> 5, lane = threadIdx.x & 31;
    int n15 = lane & 15, kh = lane >> 4;
    int tile = blockIdx.x * 8 + wave;
    if (tile >= (NN / 16)) return;   // uniform per wave, no barriers in this kernel
    int node0 = tile * 16;
    int which = blockIdx.y;
    const unsigned short* wmat = wT + (size_t)which * 16384;
    float* out = which ? hB : hA;

    v16s a[4];
    const unsigned short* arow = hb + (size_t)(node0 + n15) * DD;
#pragma unroll
    for (int f = 0; f < 4; f++) a[f] = afrag_bf(arow, f, kh);

#pragma unroll
    for (int nt = 0; nt < 8; nt++) {
        v8f acc = {};
#pragma unroll
        for (int f = 0; f < 4; f++) acc = wmma_bf16(a[f], bfrag(wmat, nt, n15, f, kh), acc);
        float bias = which ? 0.f : be1[nt * 16 + n15];
#pragma unroll
        for (int r = 0; r < 8; r++)
            out[(size_t)(node0 + r + 8 * kh) * DD + nt * 16 + n15] = acc[r] + bias;
    }
}

// ---------------- fused edge pipeline: one wave handles 16 edges ----------------
__global__ __launch_bounds__(128) void k_edge(
    const float* __restrict__ x, const int* __restrict__ edges,
    const float* __restrict__ hA, const float* __restrict__ hB,
    const float* __restrict__ we1,   // for row 256 (radial weights)
    const float* __restrict__ be2, const float* __restrict__ bc1,
    const float* __restrict__ wc2,
    const unsigned short* __restrict__ wT,
    float* __restrict__ agg_m, float* __restrict__ agg_x, float* __restrict__ cnt) {
    __shared__ float lds[4][16][132];
    int wave = threadIdx.x >> 5, lane = threadIdx.x & 31;
    int n15 = lane & 15, kh = lane >> 4;
    int tile = blockIdx.x * 4 + wave;
    int base = tile * 16;

    int eM = base + n15; if (eM >= NE) eM = NE - 1;
    int row = edges[eM], col = edges[NE + eM];
    float cdx = x[row * 3 + 0] - x[col * 3 + 0];
    float cdy = x[row * 3 + 1] - x[col * 3 + 1];
    float cdz = x[row * 3 + 2] - x[col * 3 + 2];
    float radial = cdx * cdx + cdy * cdy + cdz * cdz;

    // broadcast row indices of all 16 edges (lane M holds edge base+M)
    int rM[8];
#pragma unroll
    for (int r = 0; r < 8; r++) rM[r] = __shfl(row, r + 8 * kh, 32);

    // first edge layer (gather form): t = silu(hA[row] + hB[col] + radial*we1c)  -> A frags
    const float* we1c = we1 + 256 * DD;
    const float* pa = hA + (size_t)row * DD;
    const float* pb = hB + (size_t)col * DD;
    v16s a[4];
#pragma unroll
    for (int f = 0; f < 4; f++) {
        int k0 = f * 32 + kh * 8;
#pragma unroll
        for (int half = 0; half < 2; half++) {
            int kk = k0 + half * 16;
            float ta[8], tb[8], tc[8];
            ld8(pa + kk, ta); ld8(pb + kk, tb); ld8(we1c + kk, tc);
#pragma unroll
            for (int i = 0; i < 8; i++)
                a[f][half * 8 + i] = (short)f2bf(silu(ta[i] + tb[i] + radial * tc[i]));
        }
    }

    const unsigned short* we2T = wT + 2 * 16384;
    const unsigned short* wc1T = wT + 3 * 16384;

    // GEMM1: m = silu(t @ we2 + be2); scatter into agg_m + stage in LDS for relayout
#pragma unroll
    for (int nt = 0; nt < 8; nt++) {
        v8f acc = {};
#pragma unroll
        for (int f = 0; f < 4; f++) acc = wmma_bf16(a[f], bfrag(we2T, nt, n15, f, kh), acc);
        float bias = be2[nt * 16 + n15];
#pragma unroll
        for (int r = 0; r < 8; r++) {
            float mv = silu(acc[r] + bias);
            lds[wave][r + 8 * kh][nt * 16 + n15] = mv;
            if (base + r + 8 * kh < NE)
                atomicAdd(agg_m + (size_t)rM[r] * DD + nt * 16 + n15, mv);
        }
    }
    __syncthreads();   // executed uniformly by all 4 waves

    // reload m as A-fragments
    v16s ma[4];
#pragma unroll
    for (int f = 0; f < 4; f++) {
        int k0 = f * 32 + kh * 8;
#pragma unroll
        for (int half = 0; half < 2; half++) {
            int kk = k0 + half * 16;
            float tm[8];
            ld8(&lds[wave][n15][kk], tm);
#pragma unroll
            for (int i = 0; i < 8; i++) ma[f][half * 8 + i] = (short)f2bf(tm[i]);
        }
    }

    // GEMM2 + fused wc2 dot: w = silu(m @ wc1 + bc1) @ wc2
    float p[8] = {0, 0, 0, 0, 0, 0, 0, 0};
#pragma unroll
    for (int nt = 0; nt < 8; nt++) {
        v8f acc = {};
#pragma unroll
        for (int f = 0; f < 4; f++) acc = wmma_bf16(ma[f], bfrag(wc1T, nt, n15, f, kh), acc);
        float bias = bc1[nt * 16 + n15];
        float wv = wc2[nt * 16 + n15];
#pragma unroll
        for (int r = 0; r < 8; r++) p[r] += silu(acc[r] + bias) * wv;
    }
#pragma unroll
    for (int r = 0; r < 8; r++) {
        p[r] += __shfl_xor(p[r], 1, 32);
        p[r] += __shfl_xor(p[r], 2, 32);
        p[r] += __shfl_xor(p[r], 4, 32);
        p[r] += __shfl_xor(p[r], 8, 32);
    }

    // coord update: trans = clip(coord_diff * w), aggregated by row (+count)
#pragma unroll
    for (int r = 0; r < 8; r++) {
        int src = r + 8 * kh;
        float wx = __shfl(cdx, src, 32) * p[r];
        float wy = __shfl(cdy, src, 32) * p[r];
        float wz = __shfl(cdz, src, 32) * p[r];
        wx = fminf(fmaxf(wx, -CLAMPV), CLAMPV);
        wy = fminf(fmaxf(wy, -CLAMPV), CLAMPV);
        wz = fminf(fmaxf(wz, -CLAMPV), CLAMPV);
        if (n15 == 0 && (base + src) < NE) {
            atomicAdd(agg_x + (size_t)rM[r] * 3 + 0, wx);
            atomicAdd(agg_x + (size_t)rM[r] * 3 + 1, wy);
            atomicAdd(agg_x + (size_t)rM[r] * 3 + 2, wz);
            atomicAdd(cnt + rM[r], 1.0f);
        }
    }
}

// ---------------- node MLP: h_out = h + silu([h,agg_m]@wn1+bn1)@wn2 + bn2 ----------------
__global__ __launch_bounds__(128) void k_node(
    const float* __restrict__ h, const unsigned short* __restrict__ hb,
    const float* __restrict__ agg_m,
    const float* __restrict__ bn1, const float* __restrict__ bn2,
    const unsigned short* __restrict__ wT,
    float* __restrict__ hout) {
    __shared__ float lds[4][16][132];
    int wave = threadIdx.x >> 5, lane = threadIdx.x & 31;
    int n15 = lane & 15, kh = lane >> 4;
    int tile = blockIdx.x * 4 + wave;
    bool valid = tile < (NN / 16);
    if (!valid) tile = (NN / 16) - 1;     // clamp; keep barriers uniform
    int node0 = tile * 16;

    const unsigned short* wn1aT = wT + 4 * 16384;
    const unsigned short* wn1bT = wT + 5 * 16384;
    const unsigned short* wn2T  = wT + 6 * 16384;

    v16s a1[4], a2[4];
    const unsigned short* hrow = hb + (size_t)(node0 + n15) * DD;
    const float* grow = agg_m + (size_t)(node0 + n15) * DD;
#pragma unroll
    for (int f = 0; f < 4; f++) {
        a1[f] = afrag_bf(hrow, f, kh);
        int k0 = f * 32 + kh * 8;
#pragma unroll
        for (int half = 0; half < 2; half++) {
            int kk = k0 + half * 16;
            float tg[8]; ld8(grow + kk, tg);
#pragma unroll
            for (int i = 0; i < 8; i++) a2[f][half * 8 + i] = (short)f2bf(tg[i]);
        }
    }

#pragma unroll
    for (int nt = 0; nt < 8; nt++) {
        v8f acc = {};
#pragma unroll
        for (int f = 0; f < 4; f++) acc = wmma_bf16(a1[f], bfrag(wn1aT, nt, n15, f, kh), acc);
#pragma unroll
        for (int f = 0; f < 4; f++) acc = wmma_bf16(a2[f], bfrag(wn1bT, nt, n15, f, kh), acc);
        float bias = bn1[nt * 16 + n15];
#pragma unroll
        for (int r = 0; r < 8; r++)
            lds[wave][r + 8 * kh][nt * 16 + n15] = silu(acc[r] + bias);
    }
    __syncthreads();

    v16s u[4];
#pragma unroll
    for (int f = 0; f < 4; f++) {
        int k0 = f * 32 + kh * 8;
#pragma unroll
        for (int half = 0; half < 2; half++) {
            int kk = k0 + half * 16;
            float tu[8];
            ld8(&lds[wave][n15][kk], tu);
#pragma unroll
            for (int i = 0; i < 8; i++) u[f][half * 8 + i] = (short)f2bf(tu[i]);
        }
    }

#pragma unroll
    for (int nt = 0; nt < 8; nt++) {
        v8f acc = {};
#pragma unroll
        for (int f = 0; f < 4; f++) acc = wmma_bf16(u[f], bfrag(wn2T, nt, n15, f, kh), acc);
        float bias = bn2[nt * 16 + n15];
        if (valid) {
#pragma unroll
            for (int r = 0; r < 8; r++) {
                size_t idx = (size_t)(node0 + r + 8 * kh) * DD + nt * 16 + n15;
                hout[idx] = h[idx] + acc[r] + bias;
            }
        }
    }
}

// ---------------- x_out = x + agg_x / max(cnt,1) ----------------
__global__ void k_xout(const float* __restrict__ x, const float* __restrict__ agg_x,
                       const float* __restrict__ cnt, float* __restrict__ xout) {
    int i = blockIdx.x * blockDim.x + threadIdx.x;
    if (i >= NN) return;
    float inv = 1.0f / fmaxf(cnt[i], 1.0f);
    xout[i * 3 + 0] = x[i * 3 + 0] + agg_x[i * 3 + 0] * inv;
    xout[i * 3 + 1] = x[i * 3 + 1] + agg_x[i * 3 + 1] * inv;
    xout[i * 3 + 2] = x[i * 3 + 2] + agg_x[i * 3 + 2] * inv;
}

extern "C" void kernel_launch(void* const* d_in, const int* in_sizes, int n_in,
                              void* d_out, int out_size, void* d_ws, size_t ws_size,
                              hipStream_t stream) {
    (void)in_sizes; (void)n_in; (void)out_size; (void)ws_size;
    const float* h   = (const float*)d_in[0];
    const float* x   = (const float*)d_in[1];
    const int* edges = (const int*)d_in[2];
    const float* we1 = (const float*)d_in[3];
    const float* be1 = (const float*)d_in[4];
    const float* we2 = (const float*)d_in[5];
    const float* be2 = (const float*)d_in[6];
    const float* wn1 = (const float*)d_in[7];
    const float* bn1 = (const float*)d_in[8];
    const float* wn2 = (const float*)d_in[9];
    const float* bn2 = (const float*)d_in[10];
    const float* wc1 = (const float*)d_in[11];
    const float* bc1 = (const float*)d_in[12];
    const float* wc2 = (const float*)d_in[13];
    (void)bn2;

    char* ws = (char*)d_ws;
    size_t off = 0;
    auto take = [&](size_t bytes) -> void* {
        void* p = ws + off;
        off += (bytes + 255) & ~(size_t)255;
        return p;
    };
    unsigned short* hb    = (unsigned short*)take((size_t)NN * DD * 2);
    float*          hA    = (float*)take((size_t)NN * DD * 4);
    float*          hB    = (float*)take((size_t)NN * DD * 4);
    float*          agg_m = (float*)take((size_t)NN * DD * 4);
    float*          agg_x = (float*)take((size_t)NN * 3 * 4);
    float*          cntb  = (float*)take((size_t)NN * 4);
    unsigned short* wT    = (unsigned short*)take((size_t)7 * 16384 * 2);

    k_prep<<<(NN * DD + 255) / 256, 256, 0, stream>>>(h, hb, agg_m, agg_x, cntb);
    dim3 gw(16384 / 256, 7);
    k_wT<<<gw, 256, 0, stream>>>(we1, we2, wc1, wn1, wn2, wT);
    dim3 gh((NN / 16 + 7) / 8, 2);
    k_hAhB<<<gh, 256, 0, stream>>>(hb, wT, be1, hA, hB);
    k_edge<<<(NE / 16) / 4, 128, 0, stream>>>(x, edges, hA, hB, we1, be2, bc1, wc2, wT,
                                              agg_m, agg_x, cntb);
    k_node<<<(NN / 16 + 3) / 4, 128, 0, stream>>>(h, hb, agg_m, bn1, (const float*)d_in[10],
                                                  wT, (float*)d_out);
    k_xout<<<(NN + 255) / 256, 256, 0, stream>>>(x, agg_x, cntb, (float*)d_out + (size_t)NN * DD);
}
